// ListenerModel_59751585022368
// MI455X (gfx1250) — matrix operand order repaired and checked by reading, not compiled, not run on hardware
//
#include <hip/hip_runtime.h>
#include <hip/hip_bf16.h>
#include <math.h>

typedef __bf16 bf16;
typedef __attribute__((ext_vector_type(16))) __bf16 v16bf;
typedef __attribute__((ext_vector_type(8)))  float  v8f;

#define V_    32000
#define E_    512
#define H_    2048
#define IMG_  2048
#define A_    512
#define B_    64
#define S_    256
#define HM_   64
#define MTEXT (B_*S_)   // 16384
#define MSEP  (B_*6)    // 384
#define CTXK  (IMG_*6)  // 12288

// ============================================================================
// WMMA GEMM: out = act( A @ Wt^T + bias + rowAdd[m/rowDiv] )
//   A: bf16 [M,K] row-major, or (GATHER) f32 rows of emb selected by gidx[m]
//   Wt: bf16 [N,K] row-major (pre-transposed weight)
// Block tile 128x128, 8 waves (4 M x 2 N), wave tile 32x64, K-step 32.
// ============================================================================
template<bool GATHER>
__global__ __launch_bounds__(256) void gemm_wmma(
    const void* __restrict__ Asrc, const int* __restrict__ gidx,
    const bf16* __restrict__ Wt,
    const float* __restrict__ bias, const float* __restrict__ rowAdd, int rowDiv,
    int M, int N, int K, int act, bf16* outB, float* outF)
{
  __shared__ bf16 As[128 * 48];   // 48-elem row stride (96B, 16B aligned)
  __shared__ bf16 Bs[128 * 48];

  const int tid  = threadIdx.x;
  const int lane = tid & 31;
  const int wid  = tid >> 5;
  const int wm   = wid >> 1;        // 0..3
  const int wn   = wid & 1;         // 0..1
  const int l15  = lane & 15;
  const int lh   = lane >> 4;       // lane half
  const int bm   = blockIdx.x * 128;
  const int bn   = blockIdx.y * 128;
  const int lrow = tid >> 1;        // 0..127  (tile row loaded by this thread)
  const int lseg = (tid & 1) * 16;  // 0 | 16  (k segment)

  v8f acc[2][4] = {};

  for (int k0 = 0; k0 < K; k0 += 32) {
    // ---- stage A tile (128 x 32) ----
    if (GATHER) {
      const int grow = gidx[bm + lrow];
      const float* src = (const float*)Asrc + (size_t)grow * K + (k0 + lseg);
      union { bf16 h[16]; uint4 u[2]; } tmp;
      #pragma unroll
      for (int v = 0; v < 4; v++) {
        float4 f = ((const float4*)src)[v];
        tmp.h[v*4+0] = (bf16)f.x; tmp.h[v*4+1] = (bf16)f.y;
        tmp.h[v*4+2] = (bf16)f.z; tmp.h[v*4+3] = (bf16)f.w;
      }
      uint4* dst = (uint4*)&As[lrow*48 + lseg];
      dst[0] = tmp.u[0]; dst[1] = tmp.u[1];
    } else {
      const bf16* src = (const bf16*)Asrc + (size_t)(bm + lrow) * K + (k0 + lseg);
      uint4* dst = (uint4*)&As[lrow*48 + lseg];
      dst[0] = ((const uint4*)src)[0]; dst[1] = ((const uint4*)src)[1];
    }
    // ---- stage B tile: rows of Wt are output columns ----
    {
      const bf16* src = Wt + (size_t)(bn + lrow) * K + (k0 + lseg);
      uint4* dst = (uint4*)&Bs[lrow*48 + lseg];
      dst[0] = ((const uint4*)src)[0]; dst[1] = ((const uint4*)src)[1];
      if (k0 + 32 < K) __builtin_prefetch(src + 32, 0, 1);  // global_prefetch_b8
    }
    __syncthreads();

    // ---- fragments (layouts per CDNA5 ISA 7.12.2) ----
    // A 16x32: lanes 0-15 hold K0-7 then K16-23; lanes 16-31 hold K8-15 then K24-31
    union { uint4 u[2]; v16bf v; } af[2], bfr[4];
    #pragma unroll
    for (int tm = 0; tm < 2; tm++) {
      const bf16* ap = &As[(wm*32 + tm*16 + l15) * 48 + lh*8];
      af[tm].u[0] = *(const uint4*)ap;          // ds_read_b128
      af[tm].u[1] = *(const uint4*)(ap + 16);
    }
    // B 32x16: lanes 0-15 hold K0-7 then K16-23; lanes 16-31 hold K8-15 then K24-31 (col = l15)
    #pragma unroll
    for (int tn = 0; tn < 4; tn++) {
      const bf16* bp = &Bs[(wn*64 + tn*16 + l15) * 48 + lh*8];
      bfr[tn].u[0] = *(const uint4*)bp;
      bfr[tn].u[1] = *(const uint4*)(bp + 16);
    }
    #pragma unroll
    for (int tm = 0; tm < 2; tm++)
      #pragma unroll
      for (int tn = 0; tn < 4; tn++)
        acc[tm][tn] = __builtin_amdgcn_wmma_f32_16x16x32_bf16(
            false, af[tm].v, false, bfr[tn].v, (short)0, acc[tm][tn], false, false);
    __syncthreads();
  }

  // ---- epilogue: C/D layout -> vgpr r = row (r | r+8 by lane half), col = l15
  #pragma unroll
  for (int tm = 0; tm < 2; tm++) {
    #pragma unroll
    for (int tn = 0; tn < 4; tn++) {
      const int n  = bn + wn*64 + tn*16 + l15;
      const int mb = bm + wm*32 + tm*16 + lh*8;
      const float bv = bias ? bias[n] : 0.0f;
      #pragma unroll
      for (int r = 0; r < 8; r++) {
        const int m = mb + r;
        float v = acc[tm][tn][r] + bv;
        if (rowAdd) v += rowAdd[(size_t)(m / rowDiv) * N + n];
        if (act) v = fmaxf(v, 0.0f);
        if (outB) outB[(size_t)m * N + n] = (bf16)v;
        else      outF[(size_t)m * N + n] = v;
      }
    }
  }
}

// ============================================================================
// Helper kernels (memory-bound glue)
// ============================================================================
__global__ void k_transpose_bf16(const float* __restrict__ W, bf16* __restrict__ Wt,
                                 int K, int N) {
  int i = blockIdx.x * 256 + threadIdx.x;
  if (i < K * N) {
    int k = i / N, n = i - k * N;
    Wt[(size_t)n * K + k] = (bf16)W[i];
  }
}

__global__ void k_convert_bf16(const float* __restrict__ x, bf16* __restrict__ y, int n) {
  int i = blockIdx.x * 256 + threadIdx.x;
  if (i < n) y[i] = (bf16)x[i];
}

// ctx[b,h] = relu( vc[b,:] @ W_ctx[:,h] + b_ctx[h] )   (64x2048, K=12288)
__global__ void k_ctx(const float* __restrict__ vc, const float* __restrict__ Wc,
                      const float* __restrict__ bc, float* __restrict__ ctx) {
  int idx = blockIdx.x * 256 + threadIdx.x;        // 64*2048
  int b = idx >> 11, h = idx & 2047;
  const float* v = vc + (size_t)b * CTXK;
  float acc = bc[h];
  for (int k = 0; k < CTXK; k++) acc = fmaf(v[k], Wc[(size_t)k * H_ + h], acc);
  ctx[idx] = fmaxf(acc, 0.0f);
}

// ctxmm[b,n] = ctx[b,:] @ W_mm[H: , n]   (bias b_mm added in GEMM2 epilogue)
__global__ void k_ctxmm(const float* __restrict__ ctx, const float* __restrict__ Wmm,
                        float* __restrict__ out) {
  int idx = blockIdx.x * 256 + threadIdx.x;        // 64*2048
  int b = idx >> 11, n = idx & 2047;
  const float* c = ctx + (size_t)b * H_;
  float acc = 0.0f;
  for (int h = 0; h < H_; h++) acc = fmaf(c[h], Wmm[(size_t)(H_ + h) * H_ + n], acc);
  out[idx] = acc;
}

// ragged-history masked average -> bf16 [384, 512]
__global__ void k_hist_avg(const int* __restrict__ toks, const int* __restrict__ lens,
                           const float* __restrict__ emb, bf16* __restrict__ havg) {
  int bj = blockIdx.x;            // 0..383
  int e  = threadIdx.x;           // 0..255, covers e and e+256
  int len = lens[bj]; if (len > HM_) len = HM_;
  float a0 = 0.0f, a1 = 0.0f;
  for (int t = 0; t < len; t++) {
    int tok = toks[bj * HM_ + t];
    const float* r = emb + (size_t)tok * E_;
    a0 += r[e]; a1 += r[e + 256];
  }
  float inv = 1.0f / (float)(len > 0 ? len : 1);
  havg[(size_t)bj * E_ + e]       = (bf16)(a0 * inv);
  havg[(size_t)bj * E_ + e + 256] = (bf16)(a1 * inv);
}

// att[i] = mask ? -1e9 : ( tanh(tpre[i,:]) @ W_a2 + b_a2 )
__global__ void k_att(const float* __restrict__ tpre, const float* __restrict__ Wa2,
                      const float* __restrict__ ba2, const unsigned char* __restrict__ masks,
                      float* __restrict__ att) {
  __shared__ float red[256];
  int i = blockIdx.x, t = threadIdx.x;
  const float* row = tpre + (size_t)i * A_;
  float p = tanhf(row[t]) * Wa2[t] + tanhf(row[t + 256]) * Wa2[t + 256];
  red[t] = p; __syncthreads();
  for (int o = 128; o > 0; o >>= 1) { if (t < o) red[t] += red[t + o]; __syncthreads(); }
  if (t == 0) att[i] = masks[i] ? -1e9f : (red[0] + ba2[0]);
}

// softmax over S per batch
__global__ void k_softmax(const float* __restrict__ att, float* __restrict__ attw) {
  __shared__ float red[256];
  int b = blockIdx.x, s = threadIdx.x;
  float v = att[b * S_ + s];
  red[s] = v; __syncthreads();
  for (int o = 128; o > 0; o >>= 1) { if (s < o) red[s] = fmaxf(red[s], red[s + o]); __syncthreads(); }
  float m = red[0]; __syncthreads();
  float e = __expf(v - m);
  red[s] = e; __syncthreads();
  for (int o = 128; o > 0; o >>= 1) { if (s < o) red[s] += red[s + o]; __syncthreads(); }
  attw[b * S_ + s] = e / red[0];
}

// attended[b,h] = sum_s mm_reps[b,s,h] * attw[b,s]
__global__ void k_attended(const bf16* __restrict__ mmreps, const float* __restrict__ attw,
                           float* __restrict__ attended) {
  int idx = blockIdx.x * 256 + threadIdx.x;        // 64*2048
  int b = idx >> 11, h = idx & 2047;
  float acc = 0.0f;
  for (int s = 0; s < S_; s++)
    acc = fmaf((float)mmreps[((size_t)(b * S_ + s)) * H_ + h], attw[b * S_ + s], acc);
  attended[idx] = acc;
}

// sep = relu(seplin + hproj*flag); out[bj] = (sep . attended[b]) / max(||sep||,1e-12)
__global__ void k_finalize(const float* __restrict__ seplin, const float* __restrict__ hproj,
                           const int* __restrict__ lens, const float* __restrict__ attended,
                           float* __restrict__ out) {
  __shared__ float r1[256], r2[256];
  int bj = blockIdx.x, t = threadIdx.x;
  int b = bj / 6;
  float flag = (lens[bj] > 0) ? 1.0f : 0.0f;
  float ss = 0.0f, dd = 0.0f;
  for (int r = 0; r < 8; r++) {
    int h = t + r * 256;
    float s = seplin[(size_t)bj * H_ + h] + hproj[(size_t)bj * H_ + h] * flag;
    s = fmaxf(s, 0.0f);
    ss += s * s;
    dd += s * attended[(size_t)b * H_ + h];
  }
  r1[t] = ss; r2[t] = dd; __syncthreads();
  for (int o = 128; o > 0; o >>= 1) {
    if (t < o) { r1[t] += r1[t + o]; r2[t] += r2[t + o]; }
    __syncthreads();
  }
  if (t == 0) out[bj] = r2[0] / fmaxf(sqrtf(r1[0]), 1e-12f);
}

// ============================================================================
extern "C" void kernel_launch(void* const* d_in, const int* in_sizes, int n_in,
                              void* d_out, int out_size, void* d_ws, size_t ws_size,
                              hipStream_t stream) {
  (void)in_sizes; (void)n_in; (void)out_size; (void)ws_size;
  const int*   text   = (const int*)  d_in[0];
  const unsigned char* masks = (const unsigned char*)d_in[1];
  const int*   htok   = (const int*)  d_in[2];
  const int*   hlen   = (const int*)  d_in[3];
  const float* sepimg = (const float*)d_in[4];
  const float* vctx   = (const float*)d_in[5];
  const float* emb    = (const float*)d_in[6];
  const float* W_sep  = (const float*)d_in[7];  const float* b_sep = (const float*)d_in[8];
  const float* W_e2h  = (const float*)d_in[9];  const float* b_e2h = (const float*)d_in[10];
  const float* W_e2H  = (const float*)d_in[11]; const float* b_e2H = (const float*)d_in[12];
  const float* W_ctx  = (const float*)d_in[13]; const float* b_ctx = (const float*)d_in[14];
  const float* W_mm   = (const float*)d_in[15]; const float* b_mm  = (const float*)d_in[16];
  const float* W_a1   = (const float*)d_in[17]; const float* b_a1  = (const float*)d_in[18];
  const float* W_a2   = (const float*)d_in[19]; const float* b_a2  = (const float*)d_in[20];
  float* out = (float*)d_out;

  // ---- workspace layout ----
  char* w = (char*)d_ws;
  size_t off = 0;
  auto take = [&](size_t bytes) { size_t o = off; off += (bytes + 255) & ~(size_t)255; return o; };
  bf16*  inputreps = (bf16*)(w + take((size_t)MTEXT * H_ * 2));
  bf16*  mmreps    = (bf16*)(w + take((size_t)MTEXT * H_ * 2));
  float* tpre      = (float*)(w + take((size_t)MTEXT * A_ * 4));
  bf16*  Wt_e2h    = (bf16*)(w + take((size_t)H_ * E_ * 2));
  bf16*  Wt_mm1    = (bf16*)(w + take((size_t)H_ * H_ * 2));
  bf16*  Wt_a1     = (bf16*)(w + take((size_t)A_ * H_ * 2));
  bf16*  Wt_sep    = (bf16*)(w + take((size_t)H_ * IMG_ * 2));
  bf16*  Wt_e2H    = (bf16*)(w + take((size_t)H_ * E_ * 2));
  float* ctx       = (float*)(w + take((size_t)B_ * H_ * 4));
  float* ctxmm     = (float*)(w + take((size_t)B_ * H_ * 4));
  bf16*  Xsep      = (bf16*)(w + take((size_t)MSEP * IMG_ * 2));
  float* seplin    = (float*)(w + take((size_t)MSEP * H_ * 4));
  bf16*  havg      = (bf16*)(w + take((size_t)MSEP * E_ * 2));
  float* hproj     = (float*)(w + take((size_t)MSEP * H_ * 4));
  float* att       = (float*)(w + take((size_t)MTEXT * 4));
  float* attw      = (float*)(w + take((size_t)MTEXT * 4));
  float* attended  = (float*)(w + take((size_t)B_ * H_ * 4));

  auto cdiv = [](int a, int b) { return (a + b - 1) / b; };

  // ---- weight prep: transpose+convert to bf16 [N,K] ----
  k_transpose_bf16<<<cdiv(E_*H_,256), 256, 0, stream>>>(W_e2h, Wt_e2h, E_, H_);
  k_transpose_bf16<<<cdiv(H_*H_,256), 256, 0, stream>>>(W_mm,  Wt_mm1, H_, H_);  // top H rows
  k_transpose_bf16<<<cdiv(H_*A_,256), 256, 0, stream>>>(W_a1,  Wt_a1,  H_, A_);
  k_transpose_bf16<<<cdiv(IMG_*H_,256), 256, 0, stream>>>(W_sep, Wt_sep, IMG_, H_);
  k_transpose_bf16<<<cdiv(E_*H_,256), 256, 0, stream>>>(W_e2H, Wt_e2H, E_, H_);
  k_convert_bf16<<<cdiv(MSEP*IMG_,256), 256, 0, stream>>>(sepimg, Xsep, MSEP * IMG_);

  // ---- ctx path (M=64, memory bound) ----
  k_ctx  <<<cdiv(B_*H_,256), 256, 0, stream>>>(vctx, W_ctx, b_ctx, ctx);
  k_ctxmm<<<cdiv(B_*H_,256), 256, 0, stream>>>(ctx, W_mm, ctxmm);

  // ---- history averaging ----
  k_hist_avg<<<MSEP, 256, 0, stream>>>(htok, hlen, emb, havg);

  // ---- GEMM1: input_reps = relu(emb[text] @ W_e2h + b)  [16384 x 2048] ----
  gemm_wmma<true><<<dim3(MTEXT/128, H_/128), 256, 0, stream>>>(
      emb, text, Wt_e2h, b_e2h, nullptr, 1, MTEXT, H_, E_, 1, inputreps, nullptr);

  // ---- GEMM2: mm_reps = relu(input_reps @ W_mm[:H] + ctxmm[b] + b_mm) ----
  gemm_wmma<false><<<dim3(MTEXT/128, H_/128), 256, 0, stream>>>(
      inputreps, nullptr, Wt_mm1, b_mm, ctxmm, S_, MTEXT, H_, H_, 1, mmreps, nullptr);

  // ---- GEMM3: tpre = mm_reps @ W_a1 + b_a1  [16384 x 512] ----
  gemm_wmma<false><<<dim3(MTEXT/128, A_/128), 256, 0, stream>>>(
      mmreps, nullptr, Wt_a1, b_a1, nullptr, 1, MTEXT, A_, H_, 0, nullptr, tpre);

  // ---- attention scores -> softmax -> attended ----
  k_att<<<MTEXT, 256, 0, stream>>>(tpre, W_a2, b_a2, masks, att);
  k_softmax<<<B_, 256, 0, stream>>>(att, attw);
  k_attended<<<cdiv(B_*H_,256), 256, 0, stream>>>(mmreps, attw, attended);

  // ---- image path GEMMs [384 rows] ----
  gemm_wmma<false><<<dim3(MSEP/128, H_/128), 256, 0, stream>>>(
      Xsep, nullptr, Wt_sep, b_sep, nullptr, 1, MSEP, H_, IMG_, 0, nullptr, seplin);
  gemm_wmma<false><<<dim3(MSEP/128, H_/128), 256, 0, stream>>>(
      havg, nullptr, Wt_e2H, b_e2H, nullptr, 1, MSEP, H_, E_, 1, nullptr, hproj);

  // ---- combine, normalize, dot ----
  k_finalize<<<MSEP, 256, 0, stream>>>(seplin, hproj, hlen, attended, out);
}
